// TransformerBlock_33148557591277
// MI455X (gfx1250) — matrix operand order repaired and checked by reading, not compiled, not run on hardware
//
#include <hip/hip_runtime.h>
#include <cstdint>
#include <cstddef>

// ---------------- constants ----------------
constexpr int Bn = 8, Cc = 128, Hh = 128, Ww = 128, HWp = Hh * Ww;
constexpr int HEADSn = 8, ECc = 256;

// ---------------- vector types ----------------
typedef __attribute__((ext_vector_type(16))) __bf16 v16bf;
typedef __attribute__((ext_vector_type(8)))  __bf16 v8bf;
typedef __attribute__((ext_vector_type(8)))  float  v8f;

union Frag  { v16bf v; v8bf h[2]; };
union B16x8 { v8bf v; uint32_t u[4]; };

__device__ inline __bf16 f2bf(float f) {
    uint32_t u = __builtin_bit_cast(uint32_t, f);
    uint32_t r = (u + 0x7FFFu + ((u >> 16) & 1u)) >> 16;
    uint16_t s = (uint16_t)r;
    return __builtin_bit_cast(__bf16, s);
}
__device__ inline float bf2f(__bf16 h) {
    uint16_t s = __builtin_bit_cast(uint16_t, h);
    uint32_t u = ((uint32_t)s) << 16;
    return __builtin_bit_cast(float, u);
}
__device__ inline v8f zero_v8f() {
    v8f z = {0.f, 0.f, 0.f, 0.f, 0.f, 0.f, 0.f, 0.f};
    return z;
}
__device__ inline v8bf zero_v8bf() {
    B16x8 z; z.u[0] = z.u[1] = z.u[2] = z.u[3] = 0u;
    return z.v;
}
#define WMMA_BF16(A, B, Cacc) \
    __builtin_amdgcn_wmma_f32_16x16x32_bf16(false, (A), false, (B), (short)0, (Cacc), false, false)

// Async Global -> LDS copy of 16 bytes per lane (CDNA5 GLOBAL_LOAD_ASYNC_TO_LDS_B128,
// tracked by ASYNCcnt). lds_off is the LDS byte offset (low 32 bits of generic
// shared pointer == LDS aperture offset).
__device__ inline void async_copy16(unsigned lds_off, const __bf16* gptr) {
    asm volatile("global_load_async_to_lds_b128 %0, %1, off"
                 :: "v"(lds_off), "v"((unsigned long long)(uintptr_t)gptr)
                 : "memory");
}
__device__ inline void wait_async0() {
    asm volatile("s_wait_asynccnt 0x0" ::: "memory");
}

// ============================================================
// Weight preparation
// ============================================================
// wqkv[t][o'][i] = sum_m W3[o,m,t] * W1[m,i]; o' in {q:0..127, k:128..255, v:256..383}
__global__ void combine_qkv_w_kernel(const float* __restrict__ q1, const float* __restrict__ q3,
                                     const float* __restrict__ k1, const float* __restrict__ k3,
                                     const float* __restrict__ v1, const float* __restrict__ v3,
                                     __bf16* __restrict__ wqkv) {
    int blk = blockIdx.x;          // 9*384 blocks
    int t = blk / 384, op = blk % 384;
    int which = op >> 7, o = op & 127, i = threadIdx.x;   // 128 threads
    const float* w1 = (which == 0) ? q1 : (which == 1) ? k1 : v1;
    const float* w3 = (which == 0) ? q3 : (which == 1) ? k3 : v3;
    float s = 0.f;
    for (int m = 0; m < 128; ++m)
        s += w3[((size_t)o * 128 + m) * 9 + t] * w1[(size_t)m * 128 + i];
    wqkv[((size_t)t * 384 + op) * 128 + i] = f2bf(s);
}

// wgi[t][o'][i]: o' 0..255 gate (g3,g1), 256..511 info (i3,i1); reduce over 256
__global__ void combine_gdfn_w_kernel(const float* __restrict__ g1, const float* __restrict__ g3,
                                      const float* __restrict__ i1, const float* __restrict__ i3,
                                      __bf16* __restrict__ wgi) {
    int blk = blockIdx.x;          // 9*512 blocks
    int t = blk / 512, op = blk % 512;
    int which = op >> 8, o = op & 255, i = threadIdx.x;   // 128 threads
    const float* w1 = (which == 0) ? g1 : i1;
    const float* w3 = (which == 0) ? g3 : i3;
    float s = 0.f;
    for (int m = 0; m < 256; ++m)
        s += w3[((size_t)o * 256 + m) * 9 + t] * w1[(size_t)m * 128 + i];
    wgi[((size_t)t * 512 + op) * 128 + i] = f2bf(s);
}

__global__ void f2bf_kernel(const float* __restrict__ in, __bf16* __restrict__ out, int n) {
    int i = blockIdx.x * 256 + threadIdx.x;
    if (i < n) out[i] = f2bf(in[i]);
}

// ============================================================
// LayerNorm over last axis (W) of NCHW; writes NHWC bf16
// ============================================================
__global__ __launch_bounds__(128)
void ln_w_kernel(const float* __restrict__ x, const float* __restrict__ g,
                 const float* __restrict__ bt, __bf16* __restrict__ out_nhwc) {
    int row = blockIdx.x;                   // (b*C + c)*H + h
    int w = threadIdx.x;                    // 0..127
    int h = row % Hh;
    int c = (row / Hh) % Cc;
    int b = row / (Hh * Cc);
    float v = x[(size_t)row * Ww + w];

    __shared__ float red[4];
    float s = v;
#pragma unroll
    for (int o = 16; o > 0; o >>= 1) s += __shfl_xor(s, o, 32);
    if ((threadIdx.x & 31) == 0) red[threadIdx.x >> 5] = s;
    __syncthreads();
    float mean = (red[0] + red[1] + red[2] + red[3]) * (1.f / Ww);
    __syncthreads();
    float d = v - mean;
    float sq = d * d;
#pragma unroll
    for (int o = 16; o > 0; o >>= 1) sq += __shfl_xor(sq, o, 32);
    if ((threadIdx.x & 31) == 0) red[threadIdx.x >> 5] = sq;
    __syncthreads();
    float var = (red[0] + red[1] + red[2] + red[3]) * (1.f / Ww);
    float y = d * rsqrtf(var + 1e-5f) * g[w] + bt[w];
    out_nhwc[(((size_t)b * Hh + h) * Ww + w) * Cc + c] = f2bf(y);
}

// ============================================================
// 3x3 conv as implicit GEMM (Cin=128, Mtot output channels)
// in: NHWC bf16 (Cin=128), wt: [9][Mtot][128] bf16, out: NHWC bf16 (Mtot)
// grid: (HW/64, Mtot/128, B), 256 threads (8 waves: 2 in M x 4 in N)
// Fully unrolled tap loop; double-buffered weight staging via async
// Global->LDS (ASYNCcnt) so tap t+1's copy overlaps tap t's 16 WMMAs.
// ============================================================
__global__ __launch_bounds__(256)
void conv3x3_wmma_kernel(const __bf16* __restrict__ in, const __bf16* __restrict__ wt,
                         __bf16* __restrict__ out, int Mtot) {
    constexpr int CIN = 128;
    int b = blockIdx.z;
    int mbase = blockIdx.y * 128;
    int ptile = blockIdx.x * 64;
    int h = ptile >> 7;          // /W
    int w0 = ptile & 127;        // %W (0 or 64)

    int tid = threadIdx.x;
    int wave = tid >> 5, lane = tid & 31;
    int wm = wave >> 2, wn = wave & 3;
    int half = lane >> 4, l16 = lane & 15;

    __shared__ __attribute__((aligned(16))) __bf16 smem[2][128 * CIN];   // double-buffered weights

    v8f acc[4];
#pragma unroll
    for (int a = 0; a < 4; ++a) acc[a] = zero_v8f();

    int pix = wn * 16 + l16;                 // 0..63, wave's B column
    int wpix = w0 + pix;
    const __bf16* inb = in + (size_t)b * HWp * CIN;

    unsigned lds0 = (unsigned)(uintptr_t)(&smem[0][0]);
    unsigned lds1 = (unsigned)(uintptr_t)(&smem[1][0]);

    // issue tap 0 staging
    {
        const __bf16* wtt = wt + (size_t)mbase * CIN;
#pragma unroll
        for (int it = 0; it < 8; ++it) {
            int i = tid * 8 + it * 2048;
            async_copy16(lds0 + i * 2, wtt + i);
        }
    }
    wait_async0();
    __syncthreads();

#pragma unroll
    for (int t = 0; t < 9; ++t) {
        // issue next tap's weight copy into the idle buffer (overlaps compute)
        if (t < 8) {
            unsigned dst = ((t + 1) & 1) ? lds1 : lds0;
            const __bf16* wtt = wt + ((size_t)(t + 1) * Mtot + mbase) * CIN;
#pragma unroll
            for (int it = 0; it < 8; ++it) {
                int i = tid * 8 + it * 2048;
                async_copy16(dst + i * 2, wtt + i);
            }
        }

        const __bf16* sm = &smem[t & 1][0];
        const int dy = t / 3 - 1, dx = t % 3 - 1;   // constants after unroll
        int hh = h + dy, ww = wpix + dx;
        bool ok = ((unsigned)hh < (unsigned)Hh) && ((unsigned)ww < (unsigned)Ww);
        const __bf16* bptr = inb + ((size_t)hh * Ww + ww) * CIN;

#pragma unroll
        for (int kc = 0; kc < 4; ++kc) {
            Frag fb;
            if (ok) {
                const __bf16* bp = bptr + kc * 32 + half * 16;
                fb.h[0] = *(const v8bf*)(bp);
                fb.h[1] = *(const v8bf*)(bp + 8);
            } else {
                fb.h[0] = zero_v8bf();
                fb.h[1] = zero_v8bf();
            }
            // hoist all 4 A-fragment loads so the WMMAs issue back-to-back
            Frag fa[4];
#pragma unroll
            for (int a = 0; a < 4; ++a) {
                int m = wm * 64 + a * 16 + l16;
                const __bf16* ap = sm + m * CIN + kc * 32 + half * 16;
                fa[a].h[0] = *(const v8bf*)(ap);
                fa[a].h[1] = *(const v8bf*)(ap + 8);
            }
#pragma unroll
            for (int a = 0; a < 4; ++a)
                acc[a] = WMMA_BF16(fa[a].v, fb.v, acc[a]);
        }

        if (t < 8) {
            wait_async0();      // own async copies done; barrier makes them WG-visible
            __syncthreads();
        }
    }

    // epilogue: D lane(l16)=column(pixel); vgpr r -> m = r + 8*half
    size_t obase = ((size_t)b * HWp + (size_t)h * Ww + w0 + pix) * Mtot + mbase + wm * 64 + 8 * half;
#pragma unroll
    for (int a = 0; a < 4; ++a) {
        B16x8 pk;
#pragma unroll
        for (int r = 0; r < 8; ++r) pk.v[r] = f2bf(acc[a][r]);
        *(v8bf*)(out + obase + a * 16) = pk.v;
    }
}

// ============================================================
// Attention score + softmax: one block per (b,head)
// score[d,e] = sum_p k[p,d]*q[p,e]; probsT[e][d] written (32 d slots, 16..31 = 0)
// ============================================================
__global__ __launch_bounds__(256)
void attn_score_kernel(const __bf16* __restrict__ qkv, const float* __restrict__ scale,
                       __bf16* __restrict__ probsT) {
    int bh = blockIdx.x;                    // b*HEADS + head
    int b = bh >> 3, hd = bh & 7;
    int tid = threadIdx.x;
    int wave = tid >> 5, lane = tid & 31;
    int half = lane >> 4, l16 = lane & 15;

    __shared__ __attribute__((aligned(16))) __bf16 stg[8][32][32];  // [wave][pix][q16|k16]
    __shared__ float red[8][256];

    v8f acc = zero_v8f();
    const __bf16* base = qkv + (size_t)b * HWp * 384;
    int qoff = hd * 16, koff = 128 + hd * 16;

    for (int p0 = wave * 32; p0 < HWp; p0 += 256) {
        const __bf16* pq = base + (size_t)(p0 + lane) * 384;
        *(v8bf*)&stg[wave][lane][0]  = *(const v8bf*)(pq + qoff);
        *(v8bf*)&stg[wave][lane][8]  = *(const v8bf*)(pq + qoff + 8);
        *(v8bf*)&stg[wave][lane][16] = *(const v8bf*)(pq + koff);
        *(v8bf*)&stg[wave][lane][24] = *(const v8bf*)(pq + koff + 8);

        Frag fa, fb;
#pragma unroll
        for (int j = 0; j < 8; ++j) {
            int kk = half * 16 + j;
            fa.v[j]     = stg[wave][kk][16 + l16];        // A[d][pix] = k
            fa.v[8 + j] = stg[wave][kk + 8][16 + l16];
        }
#pragma unroll
        for (int j = 0; j < 16; ++j)
            fb.v[j] = stg[wave][half * 16 + j][l16];      // B[pix][e] = q
        acc = WMMA_BF16(fa.v, fb.v, acc);
    }

    // reduce partial scores across waves
#pragma unroll
    for (int r = 0; r < 8; ++r)
        red[wave][(r + 8 * half) * 16 + l16] = acc[r];
    __syncthreads();
    float s = 0.f;
#pragma unroll
    for (int w8 = 0; w8 < 8; ++w8) s += red[w8][tid];
    __syncthreads();
    red[0][tid] = s;
    __syncthreads();

    if (tid < 16) {        // row d = tid, softmax over e
        float inv = 1.0f / scale[0];
        float row[16];
        float mx = -3.4e38f;
#pragma unroll
        for (int e = 0; e < 16; ++e) { row[e] = red[0][tid * 16 + e] * inv; mx = fmaxf(mx, row[e]); }
        float sum = 0.f;
#pragma unroll
        for (int e = 0; e < 16; ++e) { row[e] = expf(row[e] - mx); sum += row[e]; }
        float rs = 1.0f / sum;
#pragma unroll
        for (int e = 0; e < 16; ++e)
            probsT[(size_t)bh * 512 + e * 32 + tid] = f2bf(row[e] * rs);
    } else if (tid < 32) { // zero-pad d = 16..31
#pragma unroll
        for (int e = 0; e < 16; ++e)
            probsT[(size_t)bh * 512 + e * 32 + tid] = f2bf(0.f);
    }
}

// ============================================================
// attn[p, head*16+e] = sum_d v[p,d] * probs[d,e]  (K=16 zero-padded to 32)
// grid: (HW/512, B*HEADS), 256 threads; each wave: 16 pixels x 4 iters
// ============================================================
__global__ __launch_bounds__(256)
void attn_v_kernel(const __bf16* __restrict__ qkv, const __bf16* __restrict__ probsT,
                   __bf16* __restrict__ attn) {
    int bh = blockIdx.y;
    int b = bh >> 3, hd = bh & 7;
    int tid = threadIdx.x;
    int wave = tid >> 5, lane = tid & 31;
    int half = lane >> 4, l16 = lane & 15;

    __shared__ __attribute__((aligned(16))) __bf16 pb[16 * 32];  // probsT for this (b,hd)
    pb[tid] = probsT[(size_t)bh * 512 + tid];
    pb[tid + 256] = probsT[(size_t)bh * 512 + tid + 256];
    __syncthreads();

    Frag fb;
    fb.h[0] = *(const v8bf*)(pb + l16 * 32 + half * 8);
    fb.h[1] = *(const v8bf*)(pb + l16 * 32 + 16 + half * 8);

    const __bf16* vb = qkv + (size_t)b * HWp * 384 + 256 + hd * 16;
    __bf16* ob = attn + (size_t)b * HWp * Cc + hd * 16 + l16;

    for (int it = 0; it < 4; ++it) {
        int pt = blockIdx.x * 512 + it * 128 + wave * 16;
        Frag fa;
        fa.h[0] = *(const v8bf*)(vb + (size_t)(pt + l16) * 384 + half * 8);  // k=d (0..15)
        fa.h[1] = zero_v8bf();                                               // k=16..31 pad
        v8f acc = zero_v8f();
        acc = WMMA_BF16(fa.v, fb.v, acc);
#pragma unroll
        for (int r = 0; r < 8; ++r) {
            int pixel = pt + r + 8 * half;
            ob[(size_t)pixel * Cc] = f2bf(acc[r]);
        }
    }
}

// ============================================================
// 1x1 conv GEMM (Cout=128) + residual; in NHWC bf16, out NCHW fp32
// grid: (HW/64, 1, B), 256 threads. resid may alias out (same-element RMW).
// Weights staged to LDS with async Global->LDS copies.
// ============================================================
__global__ __launch_bounds__(256)
void gemm1x1_res_kernel(const __bf16* __restrict__ in, const __bf16* __restrict__ wt,
                        const float* __restrict__ resid, float* __restrict__ out, int Cin) {
    int b = blockIdx.z;
    int p0 = blockIdx.x * 64;
    int tid = threadIdx.x;
    int wave = tid >> 5, lane = tid & 31;
    int wm = wave >> 2, wn = wave & 3;
    int half = lane >> 4, l16 = lane & 15;

    __shared__ __attribute__((aligned(16))) __bf16 smem[128 * 128];
    unsigned lds0 = (unsigned)(uintptr_t)(&smem[0]);

    v8f acc[4];
#pragma unroll
    for (int a = 0; a < 4; ++a) acc[a] = zero_v8f();

    int pix = wn * 16 + l16;
    const __bf16* bbase = in + ((size_t)b * HWp + p0 + pix) * Cin;

    int nkh = Cin >> 7;
    for (int kh = 0; kh < nkh; ++kh) {
        __syncthreads();                    // previous buffer fully consumed
#pragma unroll
        for (int it = 0; it < 8; ++it) {
            int i = tid * 8 + it * 2048;
            int m = i >> 7, k = i & 127;
            async_copy16(lds0 + i * 2, wt + (size_t)m * Cin + kh * 128 + k);
        }
        wait_async0();
        __syncthreads();

#pragma unroll
        for (int kc = 0; kc < 4; ++kc) {
            Frag fb;
            const __bf16* bp = bbase + kh * 128 + kc * 32 + half * 16;
            fb.h[0] = *(const v8bf*)(bp);
            fb.h[1] = *(const v8bf*)(bp + 8);
            Frag fa[4];
#pragma unroll
            for (int a = 0; a < 4; ++a) {
                int m = wm * 64 + a * 16 + l16;
                const __bf16* ap = smem + m * 128 + kc * 32 + half * 16;
                fa[a].h[0] = *(const v8bf*)(ap);
                fa[a].h[1] = *(const v8bf*)(ap + 8);
            }
#pragma unroll
            for (int a = 0; a < 4; ++a)
                acc[a] = WMMA_BF16(fa[a].v, fb.v, acc[a]);
        }
    }

#pragma unroll
    for (int a = 0; a < 4; ++a) {
        int mb = wm * 64 + a * 16 + 8 * half;
#pragma unroll
        for (int r = 0; r < 8; ++r) {
            size_t idx = ((size_t)b * Cc + mb + r) * HWp + p0 + pix;
            out[idx] = acc[a][r] + resid[idx];
        }
    }
}

// ============================================================
// hp[p,c] = gelu(gi[p,c]) * gi[p,256+c], c in [0,256), exact gelu
// ============================================================
__global__ __launch_bounds__(256)
void gelu_mul_kernel(const __bf16* __restrict__ gi, __bf16* __restrict__ hp) {
    size_t i = (size_t)blockIdx.x * 256 + threadIdx.x;  // one thread per 8 channels
    size_t pixel = i >> 5;
    int c8 = (int)(i & 31) * 8;
    const __bf16* gp = gi + pixel * 512 + c8;
    v8bf gv = *(const v8bf*)(gp);
    v8bf iv = *(const v8bf*)(gp + 256);
    B16x8 o;
#pragma unroll
    for (int j = 0; j < 8; ++j) {
        float gx = bf2f(gv[j]);
        float ix = bf2f(iv[j]);
        float ge = 0.5f * gx * (1.0f + erff(gx * 0.70710678118654752f));
        o.v[j] = f2bf(ge * ix);
    }
    *(v8bf*)(hp + pixel * 256 + c8) = o.v;
}

// ============================================================
// Host-side launcher
// ============================================================
extern "C" void kernel_launch(void* const* d_in, const int* in_sizes, int n_in,
                              void* d_out, int out_size, void* d_ws, size_t ws_size,
                              hipStream_t stream) {
    (void)in_sizes; (void)n_in; (void)out_size; (void)ws_size;

    const float* x    = (const float*)d_in[0];
    const float* ln1g = (const float*)d_in[1];
    const float* ln1b = (const float*)d_in[2];
    const float* scl  = (const float*)d_in[3];
    const float* q1   = (const float*)d_in[4];
    const float* q3   = (const float*)d_in[5];
    const float* k1   = (const float*)d_in[6];
    const float* k3   = (const float*)d_in[7];
    const float* v1   = (const float*)d_in[8];
    const float* v3   = (const float*)d_in[9];
    const float* ow   = (const float*)d_in[10];
    const float* ln2g = (const float*)d_in[11];
    const float* ln2b = (const float*)d_in[12];
    const float* g1   = (const float*)d_in[13];
    const float* g3   = (const float*)d_in[14];
    const float* i1   = (const float*)d_in[15];
    const float* i3   = (const float*)d_in[16];
    const float* r1   = (const float*)d_in[17];
    float* out = (float*)d_out;

    char* wsb = (char*)d_ws;
    // workspace layout (regions reused once dead):
    //  [0,   32M)  xn  (later xn2)
    //  [32M, 128M) qkv ; gi later spans [32M,160M) over dead qkv+attn
    //  [128M,160M) attn
    //  [160M,224M) hp
    //  [224M, ...) weights + probsT (~2.2 MB)
    __bf16* xn   = (__bf16*)(wsb + 0);
    __bf16* qkv  = (__bf16*)(wsb + (size_t)33554432);
    __bf16* attn = (__bf16*)(wsb + (size_t)134217728);
    __bf16* gi   = (__bf16*)(wsb + (size_t)33554432);
    __bf16* hp   = (__bf16*)(wsb + (size_t)167772160);
    size_t wo = (size_t)234881024;
    __bf16* wqkv   = (__bf16*)(wsb + wo); wo += 9 * 384 * 128 * 2;
    __bf16* wgi    = (__bf16*)(wsb + wo); wo += 9 * 512 * 128 * 2;
    __bf16* owb    = (__bf16*)(wsb + wo); wo += 128 * 128 * 2;
    __bf16* r1b    = (__bf16*)(wsb + wo); wo += 128 * 256 * 2;
    __bf16* probsT = (__bf16*)(wsb + wo);

    // --- weight prep ---
    combine_qkv_w_kernel<<<9 * 384, 128, 0, stream>>>(q1, q3, k1, k3, v1, v3, wqkv);
    combine_gdfn_w_kernel<<<9 * 512, 128, 0, stream>>>(g1, g3, i1, i3, wgi);
    f2bf_kernel<<<(128 * 128 + 255) / 256, 256, 0, stream>>>(ow, owb, 128 * 128);
    f2bf_kernel<<<(128 * 256 + 255) / 256, 256, 0, stream>>>(r1, r1b, 128 * 256);

    // --- MDTA ---
    ln_w_kernel<<<Bn * Cc * Hh, 128, 0, stream>>>(x, ln1g, ln1b, xn);
    conv3x3_wmma_kernel<<<dim3(HWp / 64, 3, Bn), 256, 0, stream>>>(xn, wqkv, qkv, 384);
    attn_score_kernel<<<Bn * HEADSn, 256, 0, stream>>>(qkv, scl, probsT);
    attn_v_kernel<<<dim3(HWp / 512, Bn * HEADSn), 256, 0, stream>>>(qkv, probsT, attn);
    // x2 = conv1x1(attn, o_w) + x  -> stored in d_out
    gemm1x1_res_kernel<<<dim3(HWp / 64, 1, Bn), 256, 0, stream>>>(attn, owb, x, out, 128);

    // --- GDFN ---
    ln_w_kernel<<<Bn * Cc * Hh, 128, 0, stream>>>(out, ln2g, ln2b, xn);  // xn2 reuses xn
    conv3x3_wmma_kernel<<<dim3(HWp / 64, 4, Bn), 256, 0, stream>>>(xn, wgi, gi, 512);
    gelu_mul_kernel<<<(Bn * HWp * 256 / 8 + 255) / 256, 256, 0, stream>>>(gi, hp);
    // out = conv1x1(hp, r1) + x2, in-place residual on d_out (same-element RMW is safe)
    gemm1x1_res_kernel<<<dim3(HWp / 64, 1, Bn), 256, 0, stream>>>(hp, r1b, out, out, 256);
}
